// XNORPlusPlusConv2d_75170517614984
// MI455X (gfx1250) — 8-bit matrix operand order repaired and checked by reading, compiled, not run on hardware
//
#include <hip/hip_runtime.h>
#include <hip/hip_bf16.h>
#include <stdint.h>

#define B_    32
#define CIN   256
#define COUT  256
#define H_    56
#define W_    56
#define HW    (H_ * W_)
#define KPOS  9
#define KTOT  (CIN * KPOS)
#define KITER (KPOS * 2)

#define LDS_BUF 32768

typedef int v8i __attribute__((ext_vector_type(8)));
typedef int vi4 __attribute__((vector_size(16)));
typedef __attribute__((address_space(1))) vi4 gvi4;
typedef __attribute__((address_space(3))) vi4 svi4;

#if defined(__has_builtin)
#if __has_builtin(__builtin_amdgcn_global_load_async_to_lds_b128)
#define HAVE_ASYNC_LDS 1
#endif
#endif

#ifdef HAVE_ASYNC_LDS
#define ASYNC_B128(g, l)                                                      \
    __builtin_amdgcn_global_load_async_to_lds_b128(                           \
        (gvi4*)(uintptr_t)(g), (svi4*)(uintptr_t)(l), 0, 0)
#else
#define ASYNC_B128(g, l) (*(int4*)(l) = *(const int4*)(g))
#endif

__device__ static inline void wait_async_lds()
{
#ifdef HAVE_ASYNC_LDS
#if defined(__has_builtin) && __has_builtin(__builtin_amdgcn_s_wait_asynccnt)
    __builtin_amdgcn_s_wait_asynccnt(0);
#else
    asm volatile("s_wait_asynccnt 0" ::: "memory");
#endif
#endif
}

__global__ __launch_bounds__(256) void pack_x_kernel(const float* __restrict__ x,
                                                     int8_t* __restrict__ xq)
{
    __shared__ int8_t tile[64][68];
    const int hwBase = blockIdx.x * 64;
    const int ciBase = blockIdx.y * 64;
    const int n      = blockIdx.z;
    const int t      = threadIdx.x;

    const int ci_l = t >> 6;
    const int hw_l = t & 63;
#pragma unroll
    for (int i = 0; i < 16; ++i) {
        const int ci = ci_l + i * 4;
        const float v = x[(size_t)(n * CIN + ciBase + ci) * HW + hwBase + hw_l];
        tile[hw_l][ci] = (int8_t)((v > 0.0f) ? 1 : ((v < 0.0f) ? -1 : 0));
    }
    __syncthreads();

    const int hw_w = t >> 4;
    const int c4   = t & 15;
#pragma unroll
    for (int i = 0; i < 4; ++i) {
        const int hw = hw_w + i * 16;
        char4 val;
        val.x = tile[hw][c4 * 4 + 0];
        val.y = tile[hw][c4 * 4 + 1];
        val.z = tile[hw][c4 * 4 + 2];
        val.w = tile[hw][c4 * 4 + 3];
        *(char4*)&xq[((size_t)(n * HW + hwBase + hw)) * CIN + ciBase + c4 * 4] = val;
    }
}

__global__ __launch_bounds__(256) void pack_w_kernel(const float* __restrict__ w,
                                                     int8_t* __restrict__ wq)
{
    const int idx = blockIdx.x * 256 + threadIdx.x;
    if (idx >= COUT * KTOT / 4) return;
    const int co   = idx / (KTOT / 4);
    const int k    = (idx % (KTOT / 4)) * 4;
    const int kpos = k / CIN;
    const int ci   = k % CIN;
    char4 val;
    {
        float v0 = w[((size_t)(co * CIN + ci + 0) * KPOS) + kpos];
        float v1 = w[((size_t)(co * CIN + ci + 1) * KPOS) + kpos];
        float v2 = w[((size_t)(co * CIN + ci + 2) * KPOS) + kpos];
        float v3 = w[((size_t)(co * CIN + ci + 3) * KPOS) + kpos];
        val.x = (int8_t)((v0 > 0.0f) ? 1 : ((v0 < 0.0f) ? -1 : 0));
        val.y = (int8_t)((v1 > 0.0f) ? 1 : ((v1 < 0.0f) ? -1 : 0));
        val.z = (int8_t)((v2 > 0.0f) ? 1 : ((v2 < 0.0f) ? -1 : 0));
        val.w = (int8_t)((v3 > 0.0f) ? 1 : ((v3 < 0.0f) ? -1 : 0));
    }
    *(char4*)&wq[(size_t)co * KTOT + k] = val;
}

__global__ __launch_bounds__(256, 2) void xnor_conv_wmma(
    const int8_t* __restrict__ xq, const int8_t* __restrict__ wq,
    const float* __restrict__ o_scale, const float* __restrict__ h_scale,
    const float* __restrict__ w_scale, float* __restrict__ out)
{
    __shared__ __align__(16) int8_t smem[2 * LDS_BUF];

    const int t    = threadIdx.x;
    const int lane = t & 31;
    const int wave = t >> 5;
    const int wm   = wave & 3;
    const int wn   = wave >> 2;
    const int lo16 = lane & 15;
    const int hi   = lane >> 4;

    const int pBase = blockIdx.x * 128;
    const int nBase = blockIdx.y * 128;

    const int ldRow  = t >> 1;
    const int ldHalf = t & 1;

    const int p    = pBase + ldRow;
    const int pn   = p / HW;
    const int prem = p - pn * HW;
    const int poh  = prem / W_;
    const int pow_ = prem - poh * W_;

    const size_t wqBase = (size_t)(nBase + ldRow) * KTOT + (size_t)ldHalf * 64;

    v8i acc[2][4];
#pragma unroll
    for (int mt = 0; mt < 2; ++mt)
#pragma unroll
        for (int nt = 0; nt < 4; ++nt) {
            v8i z = {0, 0, 0, 0, 0, 0, 0, 0};
            acc[mt][nt] = z;
        }

    auto stage = [&](int kh, int kw, int kpos, int sub, int buf) {
        int8_t* dA = smem + buf * LDS_BUF + ldRow * 128 + ldHalf * 64;
        int8_t* dB = dA + 16384;
        const int8_t* sB = wq + wqBase + kpos * CIN + sub * 128;
#pragma unroll
        for (int i = 0; i < 4; ++i)
            ASYNC_B128(sB + i * 16, dB + i * 16);
        const int ih = poh + kh - 1;
        const int iw = pow_ + kw - 1;
        if (ih >= 0 && ih < H_ && iw >= 0 && iw < W_) {
            const int8_t* sA = xq + ((size_t)(pn * H_ + ih) * W_ + iw) * CIN
                                  + sub * 128 + ldHalf * 64;
#pragma unroll
            for (int i = 0; i < 4; ++i)
                ASYNC_B128(sA + i * 16, dA + i * 16);
        } else {
            const int4 z = make_int4(0, 0, 0, 0);
#pragma unroll
            for (int i = 0; i < 4; ++i)
                *(int4*)(dA + i * 16) = z;
        }
    };

    stage(0, 0, 0, 0, 0);
    wait_async_lds();
    __syncthreads();

    for (int kc = 0; kc < KITER; ++kc) {
        const int buf = kc & 1;

        if (kc + 1 < KITER) {
            const int kn    = kc + 1;
            const int kposN = kn >> 1;
            const int subN  = kn & 1;
            stage(kposN / 3, kposN % 3, kposN, subN, buf ^ 1);
        }

        const int8_t* curA = smem + buf * LDS_BUF;
        const int8_t* curB = curA + 16384;

#pragma unroll
        for (int j = 0; j < 2; ++j) {
            v8i afrag[2];
#pragma unroll
            for (int mt = 0; mt < 2; ++mt) {
                const int8_t* arow =
                    curA + (wm * 32 + mt * 16 + lo16) * 128 + j * 64 + hi * 16;
                const uint4 d0 = *(const uint4*)(arow);
                const uint4 d1 = *(const uint4*)(arow + 32);
                v8i a;
                a[0] = (int)d0.x; a[1] = (int)d0.y;
                a[2] = (int)d0.z; a[3] = (int)d0.w;
                a[4] = (int)d1.x; a[5] = (int)d1.y;
                a[6] = (int)d1.z; a[7] = (int)d1.w;
                afrag[mt] = a;
            }
            v8i bfrag[4];
#pragma unroll
            for (int nt = 0; nt < 4; ++nt) {
                const int8_t* bcol =
                    curB + (wn * 64 + nt * 16 + lo16) * 128 + j * 64 + hi * 16;
                const int4 e0 = *(const int4*)(bcol);
                const int4 e1 = *(const int4*)(bcol + 32);
                v8i b;
                b[0] = e0.x; b[1] = e0.y; b[2] = e0.z; b[3] = e0.w;
                b[4] = e1.x; b[5] = e1.y; b[6] = e1.z; b[7] = e1.w;
                bfrag[nt] = b;
            }

#pragma unroll
            for (int mt = 0; mt < 2; ++mt)
#pragma unroll
                for (int nt = 0; nt < 4; ++nt)
                    acc[mt][nt] = __builtin_amdgcn_wmma_i32_16x16x64_iu8(
                        true, afrag[mt], true, bfrag[nt], acc[mt][nt],
                        false, false);
        }

        wait_async_lds();
        __syncthreads();
    }

#pragma unroll
    for (int mt = 0; mt < 2; ++mt) {
#pragma unroll
        for (int nt = 0; nt < 4; ++nt) {
            const int co = nBase + wn * 64 + nt * 16 + lo16;
            const float os = o_scale[co];
#pragma unroll
            for (int r = 0; r < 8; ++r) {
                const int mLocal = wm * 32 + mt * 16 + hi * 8 + r;
                const int pp  = pBase + mLocal;
                const int n   = pp / HW;
                const int rem = pp - n * HW;
                const int oh  = rem / W_;
                const int ow  = rem - oh * W_;
                const float v = (float)acc[mt][nt][r] * os * h_scale[oh] * w_scale[ow];
                out[(((size_t)n * COUT + co) * H_ + oh) * W_ + ow] = v;
            }
        }
    }
}

extern "C" void kernel_launch(void* const* d_in, const int* in_sizes, int n_in,
                              void* d_out, int out_size, void* d_ws, size_t ws_size,
                              hipStream_t stream)
{
    const float* x       = (const float*)d_in[0];
    const float* w       = (const float*)d_in[1];
    const float* o_scale = (const float*)d_in[2];
    const float* h_scale = (const float*)d_in[3];
    const float* w_scale = (const float*)d_in[4];
    float* out = (float*)d_out;

    int8_t* xq = (int8_t*)d_ws;
    int8_t* wq = xq + (size_t)B_ * HW * CIN;

    pack_x_kernel<<<dim3(HW / 64, CIN / 64, B_), 256, 0, stream>>>(x, xq);
    pack_w_kernel<<<(COUT * KTOT / 4 + 255) / 256, 256, 0, stream>>>(w, wq);
    xnor_conv_wmma<<<dim3((B_ * HW) / 128, COUT / 128), 256, 0, stream>>>(
        xq, wq, o_scale, h_scale, w_scale, out);
}
